// _AdaptiveTransformerBlockV2_24300924960888
// MI455X (gfx1250) — matrix operand order repaired and checked by reading, not compiled, not run on hardware
//
#include <hip/hip_runtime.h>
#include <math.h>

// ---------- types ----------
typedef unsigned short u16;
typedef __attribute__((ext_vector_type(8)))  unsigned short us8;
typedef __attribute__((ext_vector_type(16))) unsigned short us16;
typedef __attribute__((ext_vector_type(16))) __bf16        v16bf;
typedef __attribute__((ext_vector_type(8)))  float         v8f;

__device__ __forceinline__ u16 f2bf(float f) {
  unsigned u = __float_as_uint(f);
  unsigned r = u + 0x7FFFu + ((u >> 16) & 1u);  // round-to-nearest-even
  return (u16)(r >> 16);
}
__device__ __forceinline__ float bf2f(u16 s) {
  return __uint_as_float(((unsigned)s) << 16);
}

// ---------- f32 -> bf16 conversion ----------
__global__ __launch_bounds__(256) void cvt_bf16_kernel(const float* __restrict__ in,
                                                       u16* __restrict__ out, size_t n) {
  for (size_t i = (size_t)blockIdx.x * 256 + threadIdx.x; i < n;
       i += (size_t)gridDim.x * 256)
    out[i] = f2bf(in[i]);
}

// ---------- LayerNorm (row) -> bf16 ----------
__global__ __launch_bounds__(256) void ln_kernel(const float* __restrict__ x,
                                                 const float* __restrict__ g,
                                                 const float* __restrict__ b,
                                                 u16* __restrict__ out, int D) {
  __shared__ float s1[256], s2[256];
  size_t row = blockIdx.x;
  const float* xr = x + row * (size_t)D;
  int t = threadIdx.x;
  float a = 0.f, q = 0.f;
  for (int i = t; i < D; i += 256) { float v = xr[i]; a += v; q += v * v; }
  s1[t] = a; s2[t] = q; __syncthreads();
  for (int st = 128; st > 0; st >>= 1) {
    if (t < st) { s1[t] += s1[t + st]; s2[t] += s2[t + st]; }
    __syncthreads();
  }
  float Df = (float)D;
  float mu = s1[0] / Df;
  float var = s2[0] / Df - mu * mu;
  float rs = rsqrtf(var + 1e-5f);
  for (int i = t; i < D; i += 256)
    out[row * (size_t)D + i] = f2bf((xr[i] - mu) * rs * g[i] + b[i]);
}

// ---------- row softmax (optionally causal), in place, f32 ----------
__global__ __launch_bounds__(256) void softmax_kernel(float* __restrict__ data,
                                                      int rowlen, int causalS) {
  __shared__ float sm[256];
  size_t row = blockIdx.x;
  float* p = data + row * (size_t)rowlen;
  int t = threadIdx.x;
  int L = rowlen;
  if (causalS > 0) L = (int)(row % (size_t)causalS) + 1;
  float m = -3.4e38f;
  for (int i = t; i < L; i += 256) m = fmaxf(m, p[i]);
  sm[t] = m; __syncthreads();
  for (int st = 128; st > 0; st >>= 1) {
    if (t < st) sm[t] = fmaxf(sm[t], sm[t + st]);
    __syncthreads();
  }
  m = sm[0]; __syncthreads();
  float s = 0.f;
  for (int i = t; i < rowlen; i += 256) {
    float e = (i < L) ? __expf(p[i] - m) : 0.f;
    p[i] = e; s += e;
  }
  sm[t] = s; __syncthreads();
  for (int st = 128; st > 0; st >>= 1) {
    if (t < st) sm[t] += sm[t + st];
    __syncthreads();
  }
  float inv = 1.f / sm[0];
  for (int i = t; i < rowlen; i += 256) p[i] *= inv;
}

// ---------- adaptation-conditioned MoE gate: softmax([h,adapt] @ gw + gb) ----------
__global__ __launch_bounds__(256) void gate_kernel(const u16* __restrict__ hb,
                                                   const float* __restrict__ adapt,
                                                   const float* __restrict__ gw,
                                                   const float* __restrict__ gb,
                                                   float* __restrict__ gate_out,
                                                   int D, int A, int S) {
  __shared__ float red[256 * 8];
  size_t row = blockIdx.x;
  int b = (int)(row / (size_t)S);
  int t = threadIdx.x;
  float acc[8];
#pragma unroll
  for (int e = 0; e < 8; ++e) acc[e] = 0.f;
  for (int d = t; d < D; d += 256) {
    float h = bf2f(hb[row * (size_t)D + d]);
    const float* wrow = gw + (size_t)d * 8;
#pragma unroll
    for (int e = 0; e < 8; ++e) acc[e] += h * wrow[e];
  }
  for (int a = t; a < A; a += 256) {
    float v = adapt[(size_t)b * A + a];
    const float* wrow = gw + (size_t)(D + a) * 8;
#pragma unroll
    for (int e = 0; e < 8; ++e) acc[e] += v * wrow[e];
  }
#pragma unroll
  for (int e = 0; e < 8; ++e) red[t * 8 + e] = acc[e];
  __syncthreads();
  for (int st = 128; st > 0; st >>= 1) {
    if (t < st)
#pragma unroll
      for (int e = 0; e < 8; ++e) red[t * 8 + e] += red[(t + st) * 8 + e];
    __syncthreads();
  }
  if (t == 0) {
    float lg[8], m = -3.4e38f, s = 0.f;
#pragma unroll
    for (int e = 0; e < 8; ++e) { lg[e] = red[e] + gb[e]; m = fmaxf(m, lg[e]); }
#pragma unroll
    for (int e = 0; e < 8; ++e) { lg[e] = __expf(lg[e] - m); s += lg[e]; }
    float inv = 1.f / s;
#pragma unroll
    for (int e = 0; e < 8; ++e) gate_out[row * 8 + e] = lg[e] * inv;
  }
}

// ---------- generic bf16 WMMA GEMM (double-buffered, async-to-LDS staging) ----------
// C[M,N] = epilogue(alpha * A[M,K] @ B[K,N] + bias), batched over grid.z (b,h).
// A: bf16 (or f32 if aF32), row-major, leading dim lda.
// W: bf16; if bt==0 layout [K,N] (ldw = row stride), if bt==1 layout [N,K] (W = B^T).
// Modes: 0 bf16 out, 1 f32 out, 2 f32 resid-add, 3 GELU->bf16, 4 gate-scaled accumulate.
#define BM 64
#define BN 64
#define BK 64
#define LDSS 72  // padded LDS row stride (elements); 144B rows, 16B aligned

enum { EPI_BF16 = 0, EPI_F32 = 1, EPI_RESID = 2, EPI_GELU = 3, EPI_MOE = 4 };

__global__ __launch_bounds__(128) void gemm_wmma_kernel(
    const void* __restrict__ Ap, const u16* __restrict__ Wp,
    const float* __restrict__ bias, float* __restrict__ C,
    u16* __restrict__ Cb, const float* __restrict__ resid,
    const float* __restrict__ gate, int K, int lda, int ldw, int ldc,
    long long sAb, long long sAh, long long sWb, long long sWh,
    long long sCb, long long sCh, int nh, int mode, int aF32, int bt,
    float alpha, int gateStride, int eIdx) {
  __shared__ __attribute__((aligned(16))) u16 As[2][BM * LDSS];
  __shared__ __attribute__((aligned(16))) u16 Bs[2][BN * LDSS];

  int t = threadIdx.x;
  int lane = t & 31, wid = t >> 5;
  int wy = wid >> 1, wx = wid & 1;
  int z = blockIdx.z;
  int bidx = z / nh, hidx = z % nh;

  long long aoff = (long long)bidx * sAb + (long long)hidx * sAh;
  const float* Af = (const float*)Ap + aoff;
  const u16* Ab = (const u16*)Ap + aoff;
  const u16* Wb = Wp + (long long)bidx * sWb + (long long)hidx * sWh;
  long long coff = (long long)bidx * sCb + (long long)hidx * sCh;

  int row0 = blockIdx.y * BM;
  int col0 = blockIdx.x * BN;

  v8f acc[2][2];
#pragma unroll
  for (int i = 0; i < 2; ++i)
#pragma unroll
    for (int j = 0; j < 2; ++j) acc[i][j] = 0;

  // stage one BK=64 K-tile into buffer `buf`
  auto stage = [&](int k0, int buf) {
    if (aF32) {
      // f32 source: convert while staging (synchronous path)
      if (k0 + BK < K)
        __builtin_prefetch((const void*)(Af + (size_t)(row0 + (t & 63)) * lda + k0 + BK), 0, 1);
      for (int c = t; c < 512; c += 128) {
        int row = c >> 3, col8 = (c & 7) << 3;
        const float* src = Af + (size_t)(row0 + row) * lda + k0 + col8;
        float4 f0 = *(const float4*)src;
        float4 f1 = *(const float4*)(src + 4);
        us8 val = us8{f2bf(f0.x), f2bf(f0.y), f2bf(f0.z), f2bf(f0.w),
                      f2bf(f1.x), f2bf(f1.y), f2bf(f1.z), f2bf(f1.w)};
        *(us8*)&As[buf][row * LDSS + col8] = val;
      }
    } else {
      // straight bf16 copy: CDNA5 async global->LDS (ASYNCcnt tracked)
      for (int c = t; c < 512; c += 128) {
        int row = c >> 3, col8 = (c & 7) << 3;
        unsigned lds = (unsigned)(size_t)&As[buf][row * LDSS + col8];
        const u16* g = Ab + (size_t)(row0 + row) * lda + k0 + col8;
        asm volatile("global_load_async_to_lds_b128 %0, %1, off"
                     :: "v"(lds), "v"(g) : "memory");
      }
    }
    if (bt) {
      // W is [N,K]: straight copy -> async to LDS
      for (int c = t; c < 512; c += 128) {
        int n = c >> 3, k8 = (c & 7) << 3;
        unsigned lds = (unsigned)(size_t)&Bs[buf][n * LDSS + k8];
        const u16* g = Wb + (size_t)(col0 + n) * ldw + k0 + k8;
        asm volatile("global_load_async_to_lds_b128 %0, %1, off"
                     :: "v"(lds), "v"(g) : "memory");
      }
    } else {
      // W is [K,N]: transpose while staging (synchronous path)
      for (int c = t; c < 512; c += 128) {
        int k = c >> 3, n8 = (c & 7) << 3;
        us8 v = *(const us8*)(Wb + (size_t)(k0 + k) * ldw + col0 + n8);
#pragma unroll
        for (int j = 0; j < 8; ++j) Bs[buf][(n8 + j) * LDSS + k] = v[j];
      }
    }
  };

  stage(0, 0);
  int nsteps = K / BK;
  for (int s = 0; s < nsteps; ++s) {
    int buf = s & 1;
    // my async fills done; then all waves' fills (and ds stores) visible
    asm volatile("s_wait_asynccnt 0x0" ::: "memory");
    __syncthreads();
    // overlap: fill the other buffer while computing from this one
    if (s + 1 < nsteps) stage((s + 1) * BK, buf ^ 1);

    int mlane = lane & 15;
    int kbA = (lane >> 4) << 3;   // A: 0 or 8
    int kbB = (lane >> 4) << 3;   // B: 0 or 8
#pragma unroll
    for (int kk = 0; kk < 2; ++kk) {
      union Frag { us16 w; us8 h[2]; v16bf b; } fa[2], fb[2];
#pragma unroll
      for (int i = 0; i < 2; ++i) {
        int r = (wy * 32 + i * 16 + mlane) * LDSS + kk * 32;
        fa[i].h[0] = *(const us8*)&As[buf][r + kbA];
        fa[i].h[1] = *(const us8*)&As[buf][r + 16 + kbA];
      }
#pragma unroll
      for (int j = 0; j < 2; ++j) {
        int r = (wx * 32 + j * 16 + mlane) * LDSS + kk * 32;
        fb[j].h[0] = *(const us8*)&Bs[buf][r + kbB];
        fb[j].h[1] = *(const us8*)&Bs[buf][r + 16 + kbB];
      }
#pragma unroll
      for (int i = 0; i < 2; ++i)
#pragma unroll
        for (int j = 0; j < 2; ++j)
          acc[i][j] = __builtin_amdgcn_wmma_f32_16x16x32_bf16(
              false, fa[i].b, false, fb[j].b, (short)0, acc[i][j], false, false);
    }
  }

  // ---- epilogue ----
  float* Cf = C ? C + coff : (float*)0;
  u16* Cbf = Cb ? Cb + coff : (u16*)0;
  int mhalf = (lane >> 4) << 3;
#pragma unroll
  for (int i = 0; i < 2; ++i)
#pragma unroll
    for (int j = 0; j < 2; ++j)
#pragma unroll
      for (int r = 0; r < 8; ++r) {
        int m = row0 + wy * 32 + i * 16 + mhalf + r;
        int n = col0 + wx * 32 + j * 16 + (lane & 15);
        float v = acc[i][j][r] * alpha;
        float bv = bias ? bias[n] : 0.f;
        size_t idx = (size_t)m * ldc + n;
        if (mode == EPI_BF16) {
          Cbf[idx] = f2bf(v + bv);
        } else if (mode == EPI_F32) {
          Cf[idx] = v + bv;
        } else if (mode == EPI_RESID) {
          Cf[idx] = resid[idx] + v + bv;
        } else if (mode == EPI_GELU) {
          float xg = v + bv;
          Cbf[idx] = f2bf(0.5f * xg * (1.f + erff(xg * 0.70710678118f)));
        } else {  // EPI_MOE
          float g = gate[(size_t)m * gateStride + eIdx];
          Cf[idx] = Cf[idx] + g * (v + bv);
        }
      }
}

// ---------- host ----------
extern "C" void kernel_launch(void* const* d_in, const int* in_sizes, int n_in,
                              void* d_out, int out_size, void* d_ws, size_t ws_size,
                              hipStream_t stream) {
  const int B = 2, S = 2048, SC = 256, D = 1024, Hh = 16, HC = 8, F = 4096,
            E = 8, A = 64;
  const int dh = D / Hh;    // 64
  const int dhc = D / HC;   // 128

  const float* x      = (const float*)d_in[0];
  const float* cond   = (const float*)d_in[1];
  const float* adapt  = (const float*)d_in[2];
  const float* ln1_g  = (const float*)d_in[4];
  const float* ln1_b  = (const float*)d_in[5];
  const float* ln2_g  = (const float*)d_in[6];
  const float* ln2_b  = (const float*)d_in[7];
  const float* ln3_g  = (const float*)d_in[8];
  const float* ln3_b  = (const float*)d_in[9];
  const float* sa_wq = (const float*)d_in[10]; const float* sa_bq = (const float*)d_in[11];
  const float* sa_wk = (const float*)d_in[12]; const float* sa_bk = (const float*)d_in[13];
  const float* sa_wv = (const float*)d_in[14]; const float* sa_bv = (const float*)d_in[15];
  const float* sa_wo = (const float*)d_in[16]; const float* sa_bo = (const float*)d_in[17];
  const float* ca_wq = (const float*)d_in[18]; const float* ca_bq = (const float*)d_in[19];
  const float* ca_wk = (const float*)d_in[20]; const float* ca_bk = (const float*)d_in[21];
  const float* ca_wv = (const float*)d_in[22]; const float* ca_bv = (const float*)d_in[23];
  const float* ca_wo = (const float*)d_in[24]; const float* ca_bo = (const float*)d_in[25];
  const float* gate_w = (const float*)d_in[26]; const float* gate_b = (const float*)d_in[27];
  const float* e_w1 = (const float*)d_in[28]; const float* e_b1 = (const float*)d_in[29];
  const float* e_w2 = (const float*)d_in[30]; const float* e_b2 = (const float*)d_in[31];

  // output layout: x | self_w | cross_w | gate
  float* out_x  = (float*)d_out;
  float* out_sw = out_x + (size_t)B * S * D;
  float* out_cw = out_sw + (size_t)B * Hh * S * S;
  float* out_g  = out_cw + (size_t)B * HC * S * SC;

  // workspace bump allocator
  size_t off = 0;
  char* W = (char*)d_ws;
  auto alloc = [&](size_t bytes) -> char* {
    char* p = W + off;
    off = off + ((bytes + 255) & ~(size_t)255);
    return p;
  };
  u16* wq_bf  = (u16*)alloc((size_t)D * D * 2);
  u16* wk_bf  = (u16*)alloc((size_t)D * D * 2);
  u16* wv_bf  = (u16*)alloc((size_t)D * D * 2);
  u16* wo_bf  = (u16*)alloc((size_t)D * D * 2);
  u16* cwq_bf = (u16*)alloc((size_t)D * D * 2);
  u16* cwk_bf = (u16*)alloc((size_t)D * D * 2);
  u16* cwv_bf = (u16*)alloc((size_t)D * D * 2);
  u16* cwo_bf = (u16*)alloc((size_t)D * D * 2);
  u16* ew1_bf = (u16*)alloc((size_t)E * D * F * 2);
  u16* ew2_bf = (u16*)alloc((size_t)E * F * D * 2);
  u16* cb   = (u16*)alloc((size_t)B * SC * D * 2);
  u16* hb   = (u16*)alloc((size_t)B * S * D * 2);
  u16* qb   = (u16*)alloc((size_t)B * S * D * 2);
  u16* kb   = (u16*)alloc((size_t)B * S * D * 2);
  u16* vb   = (u16*)alloc((size_t)B * S * D * 2);
  u16* ctx  = (u16*)alloc((size_t)B * S * D * 2);
  u16* kb2  = (u16*)alloc((size_t)B * SC * D * 2);
  u16* vb2  = (u16*)alloc((size_t)B * SC * D * 2);
  u16* ctx2 = (u16*)alloc((size_t)B * S * D * 2);
  u16* h1   = (u16*)alloc((size_t)B * S * F * 2);
  float* x1 = (float*)alloc((size_t)B * S * D * 4);
  float* x2 = (float*)alloc((size_t)B * S * D * 4);
  (void)ws_size; (void)n_in; (void)in_sizes; (void)out_size;

  auto cvt = [&](const float* src, u16* dst, size_t n) {
    int blocks = (int)((n + 255) / 256);
    if (blocks > 8192) blocks = 8192;
    cvt_bf16_kernel<<<blocks, 256, 0, stream>>>(src, dst, n);
  };
  auto gemm = [&](const void* Ap, const void* Wt, const float* bias, float* C,
                  u16* Cb, const float* resid, const float* gate, int M, int N,
                  int K, int lda, int ldw, int ldc, long long sAb, long long sAh,
                  long long sWb, long long sWh, long long sCb, long long sCh,
                  int nh, int Z, int mode, int aF32, int bt, float alpha,
                  int gstride, int eidx) {
    dim3 gr(N / BN, M / BM, Z);
    gemm_wmma_kernel<<<gr, 128, 0, stream>>>(
        Ap, (const u16*)Wt, bias, C, Cb, resid, gate, K, lda, ldw, ldc, sAb,
        sAh, sWb, sWh, sCb, sCh, nh, mode, aF32, bt, alpha, gstride, eidx);
  };

  const int M = B * S;  // 4096 token rows

  // 0) weights + conditioning to bf16
  cvt(sa_wq, wq_bf, (size_t)D * D);  cvt(sa_wk, wk_bf, (size_t)D * D);
  cvt(sa_wv, wv_bf, (size_t)D * D);  cvt(sa_wo, wo_bf, (size_t)D * D);
  cvt(ca_wq, cwq_bf, (size_t)D * D); cvt(ca_wk, cwk_bf, (size_t)D * D);
  cvt(ca_wv, cwv_bf, (size_t)D * D); cvt(ca_wo, cwo_bf, (size_t)D * D);
  cvt(e_w1, ew1_bf, (size_t)E * D * F);
  cvt(e_w2, ew2_bf, (size_t)E * F * D);
  cvt(cond, cb, (size_t)B * SC * D);

  // 1) self-attention
  ln_kernel<<<M, 256, 0, stream>>>(x, ln1_g, ln1_b, hb, D);
  gemm(hb, wq_bf, sa_bq, 0, qb, 0, 0, M, D, D, D, D, D, 0,0,0,0,0,0, 1, 1, EPI_BF16, 0, 0, 1.f, 0, 0);
  gemm(hb, wk_bf, sa_bk, 0, kb, 0, 0, M, D, D, D, D, D, 0,0,0,0,0,0, 1, 1, EPI_BF16, 0, 0, 1.f, 0, 0);
  gemm(hb, wv_bf, sa_bv, 0, vb, 0, 0, M, D, D, D, D, D, 0,0,0,0,0,0, 1, 1, EPI_BF16, 0, 0, 1.f, 0, 0);
  // scores[b,h] = q_h @ k_h^T / sqrt(dh)   (A async, B async via bt)
  gemm(qb, kb, 0, out_sw, 0, 0, 0, S, S, dh, D, D, S,
       (long long)S * D, dh, (long long)S * D, dh,
       (long long)Hh * S * S, (long long)S * S, Hh, B * Hh,
       EPI_F32, 0, 1, rsqrtf((float)dh), 0, 0);
  softmax_kernel<<<B * Hh * S, 256, 0, stream>>>(out_sw, S, S);  // causal
  // ctx[b,h] = probs @ v_h
  gemm(out_sw, vb, 0, 0, ctx, 0, 0, S, dh, S, S, D, D,
       (long long)Hh * S * S, (long long)S * S, (long long)S * D, dh,
       (long long)S * D, dh, Hh, B * Hh, EPI_BF16, 1, 0, 1.f, 0, 0);
  gemm(ctx, wo_bf, sa_bo, x1, 0, x, 0, M, D, D, D, D, D, 0,0,0,0,0,0, 1, 1, EPI_RESID, 0, 0, 1.f, 0, 0);

  // 2) cross-attention
  ln_kernel<<<M, 256, 0, stream>>>(x1, ln2_g, ln2_b, hb, D);
  gemm(hb, cwq_bf, ca_bq, 0, qb, 0, 0, M, D, D, D, D, D, 0,0,0,0,0,0, 1, 1, EPI_BF16, 0, 0, 1.f, 0, 0);
  gemm(cb, cwk_bf, ca_bk, 0, kb2, 0, 0, B * SC, D, D, D, D, D, 0,0,0,0,0,0, 1, 1, EPI_BF16, 0, 0, 1.f, 0, 0);
  gemm(cb, cwv_bf, ca_bv, 0, vb2, 0, 0, B * SC, D, D, D, D, D, 0,0,0,0,0,0, 1, 1, EPI_BF16, 0, 0, 1.f, 0, 0);
  gemm(qb, kb2, 0, out_cw, 0, 0, 0, S, SC, dhc, D, D, SC,
       (long long)S * D, dhc, (long long)SC * D, dhc,
       (long long)HC * S * SC, (long long)S * SC, HC, B * HC,
       EPI_F32, 0, 1, rsqrtf((float)dhc), 0, 0);
  softmax_kernel<<<B * HC * S, 256, 0, stream>>>(out_cw, SC, 0);  // no mask
  gemm(out_cw, vb2, 0, 0, ctx2, 0, 0, S, dhc, SC, SC, D, D,
       (long long)HC * S * SC, (long long)S * SC, (long long)SC * D, dhc,
       (long long)S * D, dhc, HC, B * HC, EPI_BF16, 1, 0, 1.f, 0, 0);
  gemm(ctx2, cwo_bf, ca_bo, x2, 0, x1, 0, M, D, D, D, D, D, 0,0,0,0,0,0, 1, 1, EPI_RESID, 0, 0, 1.f, 0, 0);

  // 3) MoE FFN with adaptation-conditioned gate
  ln_kernel<<<M, 256, 0, stream>>>(x2, ln3_g, ln3_b, hb, D);
  gate_kernel<<<M, 256, 0, stream>>>(hb, adapt, gate_w, gate_b, out_g, D, A, S);
  hipMemcpyAsync(out_x, x2, (size_t)B * S * D * 4, hipMemcpyDeviceToDevice, stream);
  for (int e = 0; e < E; ++e) {
    gemm(hb, ew1_bf + (size_t)e * D * F, e_b1 + (size_t)e * F, 0, h1, 0, 0,
         M, F, D, D, F, F, 0,0,0,0,0,0, 1, 1, EPI_GELU, 0, 0, 1.f, 0, 0);
    gemm(h1, ew2_bf + (size_t)e * F * D, e_b2 + (size_t)e * D, out_x, 0, 0,
         out_g, M, D, F, F, D, D, 0,0,0,0,0,0, 1, 1, EPI_MOE, 0, 0, 1.f, E, e);
  }
}
